// PhaseNN_32933809226132
// MI455X (gfx1250) — compile-verified, matrix operand order checked by reading, not run on hardware
//
#include <hip/hip_runtime.h>
#include <hip/hip_bf16.h>

// CDNA5 / gfx1250: wave32, v_wmma_f32_16x16x32_bf16 (f32 accumulate).
typedef __bf16 bf16;
typedef __attribute__((ext_vector_type(16))) __bf16 v16bf;
typedef __attribute__((ext_vector_type(8)))  __bf16 v8bf;
typedef __attribute__((ext_vector_type(8)))  float   v8f;

#define IMG    400              // 20*20
#define NBATCH 512
#define BNCNT  (NBATCH * IMG)   // per-channel BN reduction count

// ==================== prep kernels ====================

// Layer-1 input: NCHW f32 (512,2,20,20) -> NHWC bf16 (512,400,32), channels 2..31 zero.
__global__ void k_cvt_in(const float* __restrict__ F, bf16* __restrict__ y) {
  long i = (long)blockIdx.x * blockDim.x + threadIdx.x;   // over 512*400*32
  if (i < 512L * 400 * 32) {
    int  ci  = (int)(i & 31);
    long pix = i >> 5;                                    // img*400 + p
    long img = pix / IMG, p = pix - img * IMG;
    y[i] = (ci < 2) ? (bf16)F[(img * 2 + ci) * IMG + p] : (bf16)0.0f;
  }
}

// Conv weights (Co,Ci,3,3) f32 -> fragment-native chunks [k'/32][Co][32] bf16,
// k' = (kh*3+kw)*Cip + ci, zero for ci >= Ci (Cip = Ci padded to mult of 32).
__global__ void k_prep_wconv(const float* __restrict__ W, bf16* __restrict__ wb,
                             int Co, int Ci, int Cip) {
  long tot = (long)9 * Cip * Co;
  long i = (long)blockIdx.x * blockDim.x + threadIdx.x;
  if (i < tot) {
    int  kr    = (int)(i & 31);
    long t     = i >> 5;
    int  co    = (int)(t % Co);
    int  chunk = (int)(t / Co);
    int  kp    = chunk * 32 + kr;
    int  rr    = kp / Cip, ci = kp - rr * Cip;
    wb[i] = (ci < Ci) ? (bf16)W[((long)co * Ci + ci) * 9 + rr] : (bf16)0.0f;
  }
}

// Wo (200,12800) f32 -> chunks [400][208][32] bf16, rows 200..207 zero.
__global__ void k_prep_wo(const float* __restrict__ Wo, bf16* __restrict__ wb) {
  long i = (long)blockIdx.x * blockDim.x + threadIdx.x;   // over 400*208*32
  if (i < 400L * 208 * 32) {
    int  kr = (int)(i & 31);
    long t  = i >> 5;
    int  no = (int)(t % 208), chunk = (int)(t / 208);
    wb[i] = (no < 200) ? (bf16)Wo[(long)no * 12800 + chunk * 32 + kr] : (bf16)0.0f;
  }
}

__global__ void k_zero_f32(float* __restrict__ p, int n) {
  int i = blockIdx.x * blockDim.x + threadIdx.x;
  if (i < n) p[i] = 0.0f;
}

// ==================== implicit-GEMM conv + fused BN stats ====================
// out[m, co] = sum_{k'} A[m,k'] W[k',co];  m = img*400 + h*20 + w.
// NHWC input => each 32-wide K-chunk = fixed (kh,kw), 32 consecutive channels:
// A fragment = two 16-byte loads (e=0..7 at khalf+e, e=8..15 at 16+khalf+e-8).
// B pre-packed per chunk => two 16-byte loads. Wave = 16x16 tile, block covers Co.
template <int Ci, int Co>
__global__ void k_conv_wmma(const bf16* __restrict__ in, const bf16* __restrict__ wb,
                            const float* __restrict__ bias, float* __restrict__ out,
                            float* __restrict__ ssum, float* __restrict__ ssq) {
  const int lane  = threadIdx.x;
  const int wid   = threadIdx.y;
  const int m0    = blockIdx.x * 16;
  const int nloc  = lane & 15;
  const int khalf = (lane >> 4) * 8;              // 16-bit fragment layout (ISA 7.12.2)
  const int co    = wid * 16 + nloc;

  const int img = m0 / IMG;                       // 400 % 16 == 0: tile stays in one image
  const int pA  = (m0 - img * IMG) + nloc;
  const int hA  = pA / 20, wA = pA - (pA / 20) * 20;
  const bf16* __restrict__ inb = in + (size_t)img * IMG * Ci;   // [400][Ci]
  const bf16* __restrict__ bp  = wb + (size_t)co * 32 + khalf;  // chunk stride Co*32

  constexpr int NCH = Ci / 32;
  v8f acc = {};
  int chunk = 0;
  #pragma unroll
  for (int r = 0; r < 9; ++r) {
    const int  ih = hA + r / 3 - 1;
    const int  iw = wA + (r - (r / 3) * 3) - 1;
    const bool ok = ((unsigned)ih < 20u) && ((unsigned)iw < 20u);
    const bf16* __restrict__ ap = inb + (size_t)(ih * 20 + iw) * Ci + khalf;
    #pragma unroll
    for (int cc = 0; cc < NCH; ++cc, ++chunk) {
      v8bf alo = {}, ahi = {};
      if (ok) {
        alo = *(const v8bf*)(ap + cc * 32);
        ahi = *(const v8bf*)(ap + cc * 32 + 16);
      }
      const bf16* __restrict__ bpc = bp + (size_t)chunk * Co * 32;
      v8bf blo = *(const v8bf*)(bpc);
      v8bf bhi = *(const v8bf*)(bpc + 16);
      v16bf a, b;
      #pragma unroll
      for (int e = 0; e < 8; ++e) {
        a[e] = alo[e]; a[8 + e] = ahi[e];
        b[e] = blo[e]; b[8 + e] = bhi[e];
      }
      acc = __builtin_amdgcn_wmma_f32_16x16x32_bf16(false, a, false, b, (short)0, acc,
                                                    false, false);
    }
  }

  // D layout: lane holds column co, rows m0 + rbase + v. Store NHWC f32 + BN stats.
  const int   rbase = (lane < 16) ? 0 : 8;
  const int   p0    = (m0 - img * IMG) + rbase;
  const float bia   = bias[co];
  float s = 0.0f, s2 = 0.0f;
  #pragma unroll
  for (int v = 0; v < 8; ++v) {
    const float y = acc[v] + bia;
    out[((size_t)img * IMG + p0 + v) * Co + co] = y;
    s += y; s2 += y * y;
  }
  s  += __shfl_xor(s, 16);                        // lane^16 holds same column
  s2 += __shfl_xor(s2, 16);
  if (lane < 16) {
    atomicAdd(ssum + co, s);
    atomicAdd(ssq  + co, s2);
  }
}

__global__ void k_bn_finalize(const float* __restrict__ sum, const float* __restrict__ sq,
                              const float* __restrict__ g, const float* __restrict__ be,
                              float* __restrict__ scale, float* __restrict__ shift, int Co) {
  int c = threadIdx.x;
  if (c < Co) {
    float m  = sum[c] * (1.0f / BNCNT);
    float v  = sq[c] * (1.0f / BNCNT) - m * m;
    float sc = g[c] * rsqrtf(v + 1e-5f);
    scale[c] = sc;
    shift[c] = be[c] - m * sc;
  }
}

// BN+ReLU over NHWC f32 -> bf16. to_nchw=1 (layer 4) emits NCHW-flat for the FC.
__global__ void k_bn_apply_relu(const float* __restrict__ x, bf16* __restrict__ y,
                                const float* __restrict__ scale,
                                const float* __restrict__ shift,
                                int Co, long total, int to_nchw) {
  long i = (long)blockIdx.x * blockDim.x + threadIdx.x;
  if (i < total) {
    int   c  = (int)(i % Co);
    float v  = fmaf(x[i], scale[c], shift[c]);
    float rv = fmaxf(v, 0.0f);
    if (to_nchw) {
      long pix = i / Co;
      long img = pix / IMG, p = pix - img * IMG;
      y[(img * Co + c) * IMG + p] = (bf16)rv;
    } else {
      y[i] = (bf16)rv;
    }
  }
}

// ==================== FC: (512 x 12800) @ chunked bf16 [400][208][32] ====================
__global__ void k_fc_wmma(const bf16* __restrict__ A, const bf16* __restrict__ Wb,
                          const float* __restrict__ bo, float* __restrict__ out) {
  const int lane  = threadIdx.x;
  const int nt    = blockIdx.x;        // 0..12 (N padded to 208)
  const int mt    = blockIdx.y;        // 0..31
  const int nloc  = lane & 15;
  const int khalf = (lane >> 4) * 8;
  const int no    = nt * 16 + nloc;
  const int m     = mt * 16 + nloc;
  const bf16* __restrict__ arow = A + (size_t)m * 12800 + khalf;
  const bf16* __restrict__ brow = Wb + (size_t)no * 32 + khalf;   // chunk stride 208*32

  v8f acc = {};
  for (int ch = 0; ch < 400; ++ch) {
    v8bf alo = *(const v8bf*)(arow + ch * 32);
    v8bf ahi = *(const v8bf*)(arow + ch * 32 + 16);
    const bf16* __restrict__ bpc = brow + (size_t)ch * 208 * 32;
    v8bf blo = *(const v8bf*)(bpc);
    v8bf bhi = *(const v8bf*)(bpc + 16);
    v16bf a, b;
    #pragma unroll
    for (int e = 0; e < 8; ++e) {
      a[e] = alo[e]; a[8 + e] = ahi[e];
      b[e] = blo[e]; b[8 + e] = bhi[e];
    }
    acc = __builtin_amdgcn_wmma_f32_16x16x32_bf16(false, a, false, b, (short)0, acc,
                                                  false, false);
  }
  const int   rbase = (lane < 16) ? 0 : 8;
  const float bias  = (no < 200) ? bo[no] : 0.0f;
  #pragma unroll
  for (int v = 0; v < 8; ++v)
    out[(size_t)(mt * 16 + rbase + v) * 208 + no] = acc[v] + bias;
}

// ==================== phase rotation + ||t2||^2 (HBM bound: ~105 MB) ====================
__global__ void k_phase_norm(const float* __restrict__ fc, const float* __restrict__ Gr,
                             const float* __restrict__ Gi, const float* __restrict__ hrr,
                             const float* __restrict__ hri, float* __restrict__ acc_out) {
  __shared__ float su[100], sv[100];
  const int n = blockIdx.x;
  const int t = threadIdx.x;            // 256 threads = 8 waves
  if (t < 100) {
    const float c  = cosf(fc[n * 208 + t]);
    const float s  = sinf(fc[n * 208 + 100 + t]);
    const float hr = hrr[n * 100 + t], hi = hri[n * 100 + t];
    su[t] = c * hr - s * hi;
    sv[t] = s * hr + c * hi;
  }
  __syncthreads();
  const int wid = t >> 5, lane = t & 31;
  float part = 0.0f;
  for (int m = wid; m < 256; m += 8) {
    const float* grp = Gr + ((size_t)n * 256 + m) * 100;
    const float* gip = Gi + ((size_t)n * 256 + m) * 100;
    float tr = 0.0f, ti = 0.0f;
    for (int d = lane; d < 100; d += 32) {
      const float gr = grp[d], gi = gip[d];
      tr += gr * su[d] - gi * sv[d];
      ti += gr * sv[d] + gi * su[d];
    }
    #pragma unroll
    for (int o = 16; o > 0; o >>= 1) { tr += __shfl_down(tr, o); ti += __shfl_down(ti, o); }
    if (lane == 0) part += tr * tr + ti * ti;
  }
  if (lane == 0) atomicAdd(acc_out, part);
}

__global__ void k_finalize(const float* __restrict__ acc, float* __restrict__ out) {
  out[0] = -acc[0] / (float)NBATCH;
}

// ==================== host side ====================

extern "C" void kernel_launch(void* const* d_in, const int* in_sizes, int n_in,
                              void* d_out, int out_size, void* d_ws, size_t ws_size,
                              hipStream_t stream) {
  (void)in_sizes; (void)n_in; (void)out_size; (void)ws_size;
  const float* F   = (const float*)d_in[0];
  const float* Gr  = (const float*)d_in[1];
  const float* Gi  = (const float*)d_in[2];
  const float* hrr = (const float*)d_in[3];
  const float* hri = (const float*)d_in[4];
  const float* W[4]  = { (const float*)d_in[5],  (const float*)d_in[9],
                         (const float*)d_in[13], (const float*)d_in[17] };
  const float* bb[4] = { (const float*)d_in[6],  (const float*)d_in[10],
                         (const float*)d_in[14], (const float*)d_in[18] };
  const float* gg[4] = { (const float*)d_in[7],  (const float*)d_in[11],
                         (const float*)d_in[15], (const float*)d_in[19] };
  const float* be[4] = { (const float*)d_in[8],  (const float*)d_in[12],
                         (const float*)d_in[16], (const float*)d_in[20] };
  const float* Wo = (const float*)d_in[21];
  const float* bo = (const float*)d_in[22];

  char* ws = (char*)d_ws;
  const size_t ACT = 512ull * 128 * 400 * 2;      // 52.4 MB bf16 ping/pong
  bf16*  act_a = (bf16*)(ws);
  bf16*  act_b = (bf16*)(ws + ACT);
  float* convo = (float*)(ws + 2 * ACT);          // 104.9 MB f32 raw conv out (NHWC)
  size_t o = 2 * ACT + 512ull * 128 * 400 * 4;
  bf16* wb1 = (bf16*)(ws + o); o += 9ull * 32  * 128 * 2;   // [9][128][32]
  bf16* wb2 = (bf16*)(ws + o); o += 9ull * 128 * 64  * 2;
  bf16* wb3 = (bf16*)(ws + o); o += 9ull * 64  * 64  * 2;
  bf16* wb4 = (bf16*)(ws + o); o += 9ull * 64  * 32  * 2;
  bf16* wo  = (bf16*)(ws + o); o += 400ull * 208 * 32 * 2;
  float* fco   = (float*)(ws + o); o += 512ull * 208 * 4;
  float* stats = (float*)(ws + o);                // 4 layers * 512 floats + accumulator

  // zero BN-stat + norm accumulators (every call: graph-safe, deterministic)
  k_zero_f32<<<(2049 + 255) / 256, 256, 0, stream>>>(stats, 2049);

  // input -> NHWC bf16, padded to 32 channels
  k_cvt_in<<<(int)((512L * 400 * 32 + 255) / 256), 256, 0, stream>>>(F, act_a);

  // weights -> fragment-native chunked bf16
  k_prep_wconv<<<(9 * 32 * 128 + 255) / 256, 256, 0, stream>>>(W[0], wb1, 128, 2, 32);
  k_prep_wconv<<<(9 * 128 * 64 + 255) / 256, 256, 0, stream>>>(W[1], wb2, 64, 128, 128);
  k_prep_wconv<<<(9 * 64 * 64 + 255) / 256, 256, 0, stream>>>(W[2], wb3, 64, 64, 64);
  k_prep_wconv<<<(9 * 64 * 32 + 255) / 256, 256, 0, stream>>>(W[3], wb4, 32, 64, 64);
  k_prep_wo<<<(int)((400L * 208 * 32 + 255) / 256), 256, 0, stream>>>(Wo, wo);

  // conv -> BN finalize -> BN+ReLU apply (NHWC bf16; layer 4 emits NCHW for the FC)
  {
    float* st = stats + 0 * 512;
    k_conv_wmma<32, 128><<<12800, dim3(32, 8), 0, stream>>>(act_a, wb1, bb[0], convo,
                                                            st, st + 128);
    k_bn_finalize<<<1, 128, 0, stream>>>(st, st + 128, gg[0], be[0], st + 256, st + 384, 128);
    long tot = 512L * 128 * 400;
    k_bn_apply_relu<<<(int)((tot + 255) / 256), 256, 0, stream>>>(convo, act_b, st + 256,
                                                                  st + 384, 128, tot, 0);
  }
  {
    float* st = stats + 1 * 512;
    k_conv_wmma<128, 64><<<12800, dim3(32, 4), 0, stream>>>(act_b, wb2, bb[1], convo,
                                                            st, st + 128);
    k_bn_finalize<<<1, 128, 0, stream>>>(st, st + 128, gg[1], be[1], st + 256, st + 384, 64);
    long tot = 512L * 64 * 400;
    k_bn_apply_relu<<<(int)((tot + 255) / 256), 256, 0, stream>>>(convo, act_a, st + 256,
                                                                  st + 384, 64, tot, 0);
  }
  {
    float* st = stats + 2 * 512;
    k_conv_wmma<64, 64><<<12800, dim3(32, 4), 0, stream>>>(act_a, wb3, bb[2], convo,
                                                           st, st + 128);
    k_bn_finalize<<<1, 128, 0, stream>>>(st, st + 128, gg[2], be[2], st + 256, st + 384, 64);
    long tot = 512L * 64 * 400;
    k_bn_apply_relu<<<(int)((tot + 255) / 256), 256, 0, stream>>>(convo, act_b, st + 256,
                                                                  st + 384, 64, tot, 0);
  }
  {
    float* st = stats + 3 * 512;
    k_conv_wmma<64, 32><<<12800, dim3(32, 2), 0, stream>>>(act_b, wb4, bb[3], convo,
                                                           st, st + 128);
    k_bn_finalize<<<1, 128, 0, stream>>>(st, st + 128, gg[3], be[3], st + 256, st + 384, 32);
    long tot = 512L * 32 * 400;
    k_bn_apply_relu<<<(int)((tot + 255) / 256), 256, 0, stream>>>(convo, act_a, st + 256,
                                                                  st + 384, 32, tot, 1);
  }

  // FC on NCHW-flat act_a (matches reference reshape order)
  k_fc_wmma<<<dim3(13, 32), 32, 0, stream>>>(act_a, wo, bo, fco);

  // phase rotation + norm accumulation -> scalar output
  k_phase_norm<<<512, 256, 0, stream>>>(fco, Gr, Gi, hrr, hri, stats + 2048);
  k_finalize<<<1, 1, 0, stream>>>(stats + 2048, (float*)d_out);
}
